// DecoderLayer_31052613550420
// MI455X (gfx1250) — matrix operand order repaired and checked by reading, not compiled, not run on hardware
//
#include <hip/hip_runtime.h>

// ---------------------------------------------------------------------------
// Decoder layer (causal MHA + residual) for MI455X / gfx1250, wave32 WMMA.
//   X:[2,2048,1024] f32, W_q/W_k/W_v/W_o:[1024,1024] f32 -> out:[2,2048,1024] f32
// Pipeline: f32->f16 convert, QKV GEMMs (+pos emb, V transposed), flash
// attention (v_wmma_f32_16x16x32_f16, fp32 online softmax), O GEMM + residual.
// CDNA5 paths: v_wmma_f32_16x16x32_f16, global_load_async_to_lds_b128 (ASYNCcnt)
// for weight-tile staging, split wait counters, wave32 fragments.
// Workspace use: ~48 MB of d_ws (f16 intermediates).
// ---------------------------------------------------------------------------

#define DMODEL 1024
#define NHEADS 16
#define SEQ    2048
#define BATCH  2
#define DK     64
#define MROWS  (BATCH * SEQ)     // 4096
#define BHEADS (BATCH * NHEADS)  // 32

typedef __attribute__((ext_vector_type(16))) _Float16 v16h;
typedef __attribute__((ext_vector_type(8)))  float    v8f;

// Assemble a 16-element f16 fragment from two 16-byte chunks.
__device__ inline v16h frag_ld(const _Float16* p0, const _Float16* p1) {
    union { uint4 u[2]; v16h v; } t;
    t.u[0] = *reinterpret_cast<const uint4*>(p0);
    t.u[1] = *reinterpret_cast<const uint4*>(p1);
    return t.v;
}

// ---------------------------------------------------------------------------
// f32 -> f16 conversion, 4-wide
// ---------------------------------------------------------------------------
struct h4s { _Float16 x, y, z, w; };

__global__ void cvt_f32_f16(const float4* __restrict__ src,
                            _Float16* __restrict__ dst, int n4) {
    int i = blockIdx.x * blockDim.x + threadIdx.x;
    if (i < n4) {
        float4 v = src[i];
        h4s o;
        o.x = (_Float16)v.x; o.y = (_Float16)v.y;
        o.z = (_Float16)v.z; o.w = (_Float16)v.w;
        reinterpret_cast<h4s*>(dst)[i] = o;
    }
}

// ---------------------------------------------------------------------------
// Y = A @ W^T  (A:[4096][1024] f16 row-major, W:[1024][1024] f16 row-major so
// W itself is Bt[n][k]). Block = 8 waves, tile 128x64, K step 32.
// B tile staged with the CDNA5 async global->LDS path (no VGPR round-trip).
// mode 0: Q out (f16, +pos, [bh][s][dk])   mode 1: K out (same)
// mode 2: V out transposed (f16, [bh][dk][s])
// mode 3: fp32 out + residual X (final projection)
// ---------------------------------------------------------------------------
__global__ __launch_bounds__(256)
void gemm_xwT(const _Float16* __restrict__ A,
              const _Float16* __restrict__ Bt,
              const float*    __restrict__ Xres,
              _Float16*       __restrict__ outH,
              float*          __restrict__ outF,
              int mode) {
    __shared__ _Float16 Bs[64 * 32];   // 4 KB staged weight tile (n x k)

    const int lane = threadIdx.x & 31;
    const int wave = threadIdx.x >> 5;
    const int hlf  = lane >> 4;        // 0/1: lane half selects K sub-chunk
    const int l16  = lane & 15;
    const int m0   = blockIdx.x * 128 + wave * 16;
    const int n0   = blockIdx.y * 64;

    // Per-thread staging chunk: 256 threads x 16B = whole 64x32 f16 tile.
    const int srow = threadIdx.x >> 2;          // 0..63
    const int scol = (threadIdx.x & 3) << 3;    // 0,8,16,24
    const unsigned ldsOff =
        (unsigned)(uintptr_t)&Bs[srow * 32 + scol];     // LDS byte address

    v8f acc[4] = {};

    for (int k0 = 0; k0 < DMODEL; k0 += 32) {
        // Async copy Bt tile rows n0..n0+63, cols k0..k0+31 straight to LDS:
        // GLOBAL_LOAD_ASYNC_TO_LDS_B128, GV mode (64-bit vaddr), ASYNCcnt.
        {
            unsigned long long ga = (unsigned long long)(uintptr_t)
                &Bt[(size_t)(n0 + srow) * DMODEL + k0 + scol];
            asm volatile("global_load_async_to_lds_b128 %0, %1, off"
                         :: "v"(ldsOff), "v"(ga)
                         : "memory");
        }
        asm volatile("s_wait_asynccnt 0x0" ::: "memory");
        __syncthreads();

        // A fragment: row m0+l16, K chunk per lane half (ISA 16-bit A layout).
        const _Float16* pa = &A[(size_t)(m0 + l16) * DMODEL + k0 + 8 * hlf];
        v16h a = frag_ld(pa, pa + 16);

#pragma unroll
        for (int j = 0; j < 4; ++j) {
            const _Float16* pb = &Bs[(j * 16 + l16) * 32 + 8 * hlf];
            v16h b = frag_ld(pb, pb + 16);
            acc[j] = __builtin_amdgcn_wmma_f32_16x16x32_f16(
                false, a, false, b, (short)0, acc[j], false, false);
        }
        __syncthreads();
    }

    // Epilogue. C layout: lane holds col n0+j*16+l16; VGPR r -> row m0+r+8*hlf.
#pragma unroll
    for (int j = 0; j < 4; ++j) {
#pragma unroll
        for (int r = 0; r < 8; ++r) {
            int m = m0 + r + 8 * hlf;
            int n = n0 + j * 16 + l16;
            float v = acc[j][r];
            if (mode == 3) {
                outF[(size_t)m * DMODEL + n] = v + Xres[(size_t)m * DMODEL + n];
            } else {
                int bb = m >> 11, s = m & (SEQ - 1);
                int hh = n >> 6,  d = n & (DK - 1);
                if (mode <= 1) {
                    // sinusoidal pos emb (row 0 is zeros; theta uses index d)
                    float p = 0.f;
                    if (s > 0) {
                        float theta = __powf(10000.f, 2.f * (float)d * (1.f / 64.f));
                        float ang = (float)s / theta;
                        p = (d & 1) ? __cosf(ang) : __sinf(ang);
                    }
                    outH[((size_t)(bb * NHEADS + hh) * SEQ + s) * DK + d] =
                        (_Float16)(v + p);
                } else {
                    // V stored transposed: [bh][dk][s]
                    outH[((size_t)(bb * NHEADS + hh) * DK + d) * SEQ + s] =
                        (_Float16)v;
                }
            }
        }
    }
}

// ---------------------------------------------------------------------------
// Flash attention, causal. One wave per (head, 16-query strip): causal trip
// counts differ per strip, so single-wave blocks need no inter-wave barriers.
// Qh/Kh: [bh][s][dk] f16 (pos already added), Vt: [bh][dk][s] f16.
// Out Ah: [b][s][h*dk + d] f16 (row-major for the final GEMM).
// ---------------------------------------------------------------------------
__global__ __launch_bounds__(32)
void flash_attn(const _Float16* __restrict__ Qh,
                const _Float16* __restrict__ Kh,
                const _Float16* __restrict__ Vt,
                _Float16*       __restrict__ Ah) {
    __shared__ _Float16 P[16 * 32];    // C-layout -> A-layout relay for probs

    const int lane = threadIdx.x;
    const int hlf  = lane >> 4;
    const int l16  = lane & 15;
    const int bh   = blockIdx.x;            // 0..31
    const int q0   = blockIdx.y * 16;       // query strip base

    const _Float16* Qb = Qh + (size_t)bh * SEQ * DK;
    const _Float16* Kb = Kh + (size_t)bh * SEQ * DK;
    const _Float16* Vb = Vt + (size_t)bh * DK * SEQ;

    // Q fragments: 16 rows x dk=64 -> two 16x32 A fragments, loaded once.
    v16h qa[2];
#pragma unroll
    for (int kk = 0; kk < 2; ++kk) {
        const _Float16* p = Qb + (size_t)(q0 + l16) * DK + kk * 32 + 8 * hlf;
        qa[kk] = frag_ld(p, p + 16);
    }

    v8f oacc[4] = {};                       // O: 16 x 64 fp32
    float mstat[8], lstat[8];               // per-row online-softmax stats;
#pragma unroll                              // VGPR r <-> row q0 + r + 8*hlf
    for (int r = 0; r < 8; ++r) { mstat[r] = -3.0e38f; lstat[r] = 0.f; }

    const float scale = 0.125f;             // 1/sqrt(64)

    for (int k0 = 0; k0 <= q0 + 15; k0 += 32) {   // causal: keys <= q0+15
        // S = Q @ K^T for 32 keys: 2 N-subtiles x 2 K-steps = 4 WMMAs.
        v8f sc[2] = {};
#pragma unroll
        for (int j = 0; j < 2; ++j) {
#pragma unroll
            for (int kk = 0; kk < 2; ++kk) {
                const _Float16* p =
                    Kb + (size_t)(k0 + j * 16 + l16) * DK + kk * 32 + 8 * hlf;
                v16h b = frag_ld(p, p + 16);
                sc[j] = __builtin_amdgcn_wmma_f32_16x16x32_f16(
                    false, qa[kk], false, b, (short)0, sc[j], false, false);
            }
        }

        // Scale, causal mask (-1e9), online softmax update.
        float pv0[8], pv1[8];
#pragma unroll
        for (int r = 0; r < 8; ++r) {
            int row = q0 + r + 8 * hlf;
            float v0 = sc[0][r] * scale + ((k0 +      l16 > row) ? -1.0e9f : 0.f);
            float v1 = sc[1][r] * scale + ((k0 + 16 + l16 > row) ? -1.0e9f : 0.f);
            float tm = fmaxf(v0, v1);       // row-max across the 16-lane half
            tm = fmaxf(tm, __shfl_xor(tm, 1, 32));
            tm = fmaxf(tm, __shfl_xor(tm, 2, 32));
            tm = fmaxf(tm, __shfl_xor(tm, 4, 32));
            tm = fmaxf(tm, __shfl_xor(tm, 8, 32));
            float mnew = fmaxf(mstat[r], tm);
            float corr = __expf(mstat[r] - mnew);
            mstat[r] = mnew;
            float e0 = __expf(v0 - mnew);
            float e1 = __expf(v1 - mnew);
            pv0[r] = e0; pv1[r] = e1;
            float ts = e0 + e1;             // row-sum
            ts += __shfl_xor(ts, 1, 32);
            ts += __shfl_xor(ts, 2, 32);
            ts += __shfl_xor(ts, 4, 32);
            ts += __shfl_xor(ts, 8, 32);
            lstat[r] = lstat[r] * corr + ts;
#pragma unroll
            for (int j2 = 0; j2 < 4; ++j2) oacc[j2][r] *= corr;
        }

        // Relayout P (C layout) -> row-major 16x32 in LDS -> A fragment.
#pragma unroll
        for (int r = 0; r < 8; ++r) {
            P[(r + 8 * hlf) * 32 + l16]      = (_Float16)pv0[r];
            P[(r + 8 * hlf) * 32 + 16 + l16] = (_Float16)pv1[r];
        }
        __syncthreads();
        const _Float16* pp = &P[l16 * 32 + 8 * hlf];
        v16h pa = frag_ld(pp, pp + 16);
        __syncthreads();

        // O += P @ V : Vt rows are contiguous in seq -> direct B fragments.
#pragma unroll
        for (int j2 = 0; j2 < 4; ++j2) {
            const _Float16* pb =
                Vb + (size_t)(j2 * 16 + l16) * SEQ + k0 + 8 * hlf;
            v16h b = frag_ld(pb, pb + 16);
            oacc[j2] = __builtin_amdgcn_wmma_f32_16x16x32_f16(
                false, pa, false, b, (short)0, oacc[j2], false, false);
        }
    }

    // Normalize and write attn in [b][s][h*64+d] layout for the O projection.
    int bb = bh >> 4, hh = bh & 15;
#pragma unroll
    for (int j2 = 0; j2 < 4; ++j2) {
#pragma unroll
        for (int r = 0; r < 8; ++r) {
            int s = q0 + r + 8 * hlf;
            float v = oacc[j2][r] / lstat[r];
            Ah[(size_t)(bb * SEQ + s) * DMODEL + hh * DK + j2 * 16 + l16] =
                (_Float16)v;
        }
    }
}

// ---------------------------------------------------------------------------
extern "C" void kernel_launch(void* const* d_in, const int* in_sizes, int n_in,
                              void* d_out, int out_size, void* d_ws, size_t ws_size,
                              hipStream_t stream) {
    const float* X  = (const float*)d_in[0];
    const float* Wq = (const float*)d_in[1];
    const float* Wk = (const float*)d_in[2];
    const float* Wv = (const float*)d_in[3];
    const float* Wo = (const float*)d_in[4];

    // Workspace carve-up (f16 elements), ~48 MB total.
    _Float16* Xh  = (_Float16*)d_ws;
    _Float16* Wqh = Xh  + (size_t)MROWS * DMODEL;
    _Float16* Wkh = Wqh + (size_t)DMODEL * DMODEL;
    _Float16* Wvh = Wkh + (size_t)DMODEL * DMODEL;
    _Float16* Woh = Wvh + (size_t)DMODEL * DMODEL;
    _Float16* Qh  = Woh + (size_t)DMODEL * DMODEL;
    _Float16* Kh  = Qh  + (size_t)BHEADS * SEQ * DK;
    _Float16* Vt  = Kh  + (size_t)BHEADS * SEQ * DK;
    _Float16* Ah  = Vt  + (size_t)BHEADS * SEQ * DK;

    // 1) fp32 -> f16 conversions.
    {
        int n4x = (MROWS * DMODEL) / 4;
        int n4w = (DMODEL * DMODEL) / 4;
        cvt_f32_f16<<<(n4x + 255) / 256, 256, 0, stream>>>((const float4*)X,  Xh,  n4x);
        cvt_f32_f16<<<(n4w + 255) / 256, 256, 0, stream>>>((const float4*)Wq, Wqh, n4w);
        cvt_f32_f16<<<(n4w + 255) / 256, 256, 0, stream>>>((const float4*)Wk, Wkh, n4w);
        cvt_f32_f16<<<(n4w + 255) / 256, 256, 0, stream>>>((const float4*)Wv, Wvh, n4w);
        cvt_f32_f16<<<(n4w + 255) / 256, 256, 0, stream>>>((const float4*)Wo, Woh, n4w);
    }

    // 2) QKV projections (pos emb fused into Q/K epilogue, V transposed).
    dim3 gg(MROWS / 128, DMODEL / 64);
    gemm_xwT<<<gg, 256, 0, stream>>>(Xh, Wqh, nullptr, Qh, nullptr, 0);
    gemm_xwT<<<gg, 256, 0, stream>>>(Xh, Wkh, nullptr, Kh, nullptr, 1);
    gemm_xwT<<<gg, 256, 0, stream>>>(Xh, Wvh, nullptr, Vt, nullptr, 2);

    // 3) Causal flash attention: 32 heads x 128 query strips, 1 wave each.
    flash_attn<<<dim3(BHEADS, SEQ / 16), 32, 0, stream>>>(Qh, Kh, Vt, Ah);

    // 4) Output projection + residual -> fp32 d_out.
    gemm_xwT<<<gg, 256, 0, stream>>>(Ah, Woh, X, nullptr, (float*)d_out, 3);
}
